// MambaModel_30605936951376
// MI455X (gfx1250) — compile-verified, matrix operand order checked by reading, not run on hardware
//
#include <hip/hip_runtime.h>
#include <hip/hip_bf16.h>
#include <cstddef>

// ---------------------------------------------------------------------------
// Model constants (match reference)
// ---------------------------------------------------------------------------
#define V_IN   32000
#define V_OUT  32000
#define DMODEL 768
#define DSTATE 16
#define NLAYER 4
#define DCONV  4
#define DINNER 1536           // EXP * D
#define DTRANK 48             // (D+15)/16
#define SEQLEN 2048
#define XPROJ_N (DTRANK + 2*DSTATE)   // 80

typedef __attribute__((ext_vector_type(16))) _Float16 v16h;
typedef __attribute__((ext_vector_type(8)))  float    v8f;

__device__ __forceinline__ float siluf(float x) {
    return x / (1.0f + __expf(-x));
}
__device__ __forceinline__ float softplusf(float x) {
    return (x > 20.0f) ? x : log1pf(__expf(x));
}

// Load one f16 fragment (16x32 A-tile or 32x16 B-tile, ISA 16-bit layout):
// lanes 0-15 take K = kb+[0..7] and kb+[16..23]; lanes 16-31 take kb+[8..15]
// and kb+[24..31]; `p` = row base + kb + half*8.
__device__ __forceinline__ v16h load_frag(const float* __restrict__ p) {
    v16h f;
#pragma unroll
    for (int i = 0; i < 8; ++i) {
        f[i]     = (_Float16)p[i];
        f[i + 8] = (_Float16)p[16 + i];
    }
    return f;
}

// Branchless zero-padded fragment load for the K tail (no EXEC divergence).
__device__ __forceinline__ v16h load_frag_tail(const float* __restrict__ row,
                                               int kbase, int K) {
    v16h f;
#pragma unroll
    for (int i = 0; i < 8; ++i) {
        const int k0 = kbase + i;
        const int k1 = kbase + 16 + i;
        const float s0 = (k0 < K) ? 1.0f : 0.0f;
        const float s1 = (k1 < K) ? 1.0f : 0.0f;
        const int c0 = (k0 < K) ? k0 : 0;
        const int c1 = (k1 < K) ? k1 : 0;
        f[i]     = (_Float16)(row[c0] * s0);
        f[i + 8] = (_Float16)(row[c1] * s1);
    }
    return f;
}

// ---------------------------------------------------------------------------
// WMMA GEMM: C[M,N] = A[M,K] * W[N,K]^T (+ optional bias / softplus)
// fp32 operands converted to f16 fragments, f32 accumulate via
// v_wmma_f32_16x16x32_f16.
// Block = 256 threads = 8 waves arranged 4(M) x 2(N); each wave register-
// tiles a 32x64 output (2 A-frags x 4 B-frags -> 8 WMMAs per K-step), so the
// block tile is 128x128. All subtile validity checks are wave-uniform.
// EPI: 0 = none, 1 = +bias, 2 = +bias then softplus
// ---------------------------------------------------------------------------
template <int EPI>
__global__ void gemm_wmma_kernel(const float* __restrict__ A, int lda,
                                 const float* __restrict__ W, int ldw,
                                 const float* __restrict__ bias,
                                 float* __restrict__ C, int ldc,
                                 int M, int N, int K)
{
    const int wave = threadIdx.x >> 5;
    const int lane = threadIdx.x & 31;
    const int wm = wave & 3;        // 0..3  (M direction, x32)
    const int wn = wave >> 2;       // 0..1  (N direction, x64)
    const int m0 = blockIdx.y * 128 + wm * 32;
    const int n0 = blockIdx.x * 128 + wn * 64;
    if (m0 >= M || n0 >= N) return;          // wave-uniform

    const int half = lane >> 4;              // K sub-group select (ISA layout)
    const int row  = lane & 15;              // M row (A) / N row (B=W)

    const bool mv1 = (m0 + 16) < M;          // second A row-tile valid (uniform)
    bool nv[4];
    const float* __restrict__ wrow[4];
#pragma unroll
    for (int j = 0; j < 4; ++j) {
        const int nj = n0 + 16 * j;
        nv[j] = nj < N;                      // wave-uniform
        wrow[j] = W + (size_t)((nv[j] ? nj : 0) + row) * ldw;
    }
    const float* __restrict__ arow0 = A + (size_t)(m0 + row) * lda;
    const float* __restrict__ arow1 = A + (size_t)((mv1 ? m0 + 16 : m0) + row) * lda;

    v8f acc[2][4] = {};

    int kb = 0;
    for (; kb + 32 <= K; kb += 32) {
        const int ko = kb + half * 8;
        const v16h a0 = load_frag(arow0 + ko);
        const v16h a1 = load_frag(arow1 + ko);
#pragma unroll
        for (int j = 0; j < 4; ++j) {
            if (nv[j]) {
                __builtin_prefetch(wrow[j] + ko + 32, 0, 0);
                const v16h b = load_frag(wrow[j] + ko);
                acc[0][j] = __builtin_amdgcn_wmma_f32_16x16x32_f16(
                                false, a0, false, b, (short)0, acc[0][j], false, false);
                if (mv1)
                    acc[1][j] = __builtin_amdgcn_wmma_f32_16x16x32_f16(
                                    false, a1, false, b, (short)0, acc[1][j], false, false);
            }
        }
    }
    if (kb < K) {                            // zero-padded tail (e.g. K=48)
        const int kbase = kb + half * 8;
        const v16h a0 = load_frag_tail(arow0, kbase, K);
        const v16h a1 = load_frag_tail(arow1, kbase, K);
#pragma unroll
        for (int j = 0; j < 4; ++j) {
            if (nv[j]) {
                const v16h b = load_frag_tail(wrow[j], kbase, K);
                acc[0][j] = __builtin_amdgcn_wmma_f32_16x16x32_f16(
                                false, a0, false, b, (short)0, acc[0][j], false, false);
                if (mv1)
                    acc[1][j] = __builtin_amdgcn_wmma_f32_16x16x32_f16(
                                    false, a1, false, b, (short)0, acc[1][j], false, false);
            }
        }
    }

    // D layout: lane 0-15 -> N=lane, M=r ; lane 16-31 -> N=lane-16, M=8+r
#pragma unroll
    for (int j = 0; j < 4; ++j) {
        if (!nv[j]) continue;
        const int n = n0 + 16 * j + row;
        const float bv = (EPI >= 1) ? bias[n] : 0.0f;
#pragma unroll
        for (int i = 0; i < 2; ++i) {
            if (i == 1 && !mv1) continue;
            const int mb = m0 + 16 * i + half * 8;
#pragma unroll
            for (int r = 0; r < 8; ++r) {
                float v = acc[i][j][r] + bv;
                if (EPI == 2) v = softplusf(v);
                C[(size_t)(mb + r) * ldc + n] = v;
            }
        }
    }
}

// ---------------------------------------------------------------------------
// Embedding gather: H[t, :] = embed[tokens[t], :]
// ---------------------------------------------------------------------------
__global__ void embed_kernel(const int* __restrict__ tok,
                             const float* __restrict__ emb,
                             float* __restrict__ H)
{
    const int t = blockIdx.x;
    const int v = tok[t];
#pragma unroll
    for (int j = 0; j < 3; ++j) {
        const int i = threadIdx.x + j * 256;
        H[(size_t)t * DMODEL + i] = emb[(size_t)v * DMODEL + i];
    }
}

// ---------------------------------------------------------------------------
// Residual add + LayerNorm: R = first ? H : R + H ; X = LN(R) * w + b
// ---------------------------------------------------------------------------
__global__ void add_ln_kernel(const float* __restrict__ Hin,
                              float* __restrict__ R,
                              const float* __restrict__ w,
                              const float* __restrict__ b,
                              float* __restrict__ X, int first)
{
    __shared__ float ssum[256];
    __shared__ float ssq[256];
    const int t = blockIdx.x;
    const float* hrow = Hin + (size_t)t * DMODEL;
    float* rrow = R + (size_t)t * DMODEL;

    float v[3];
    float lsum = 0.0f, lsq = 0.0f;
#pragma unroll
    for (int j = 0; j < 3; ++j) {
        const int i = threadIdx.x + j * 256;
        float r = hrow[i] + (first ? 0.0f : rrow[i]);
        v[j] = r;
        lsum += r;
        lsq  += r * r;
    }
    ssum[threadIdx.x] = lsum;
    ssq[threadIdx.x]  = lsq;
    __syncthreads();
    for (int off = 128; off > 0; off >>= 1) {
        if (threadIdx.x < off) {
            ssum[threadIdx.x] += ssum[threadIdx.x + off];
            ssq[threadIdx.x]  += ssq[threadIdx.x + off];
        }
        __syncthreads();
    }
    const float mu  = ssum[0] * (1.0f / DMODEL);
    const float var = ssq[0] * (1.0f / DMODEL) - mu * mu;
    const float rs  = rsqrtf(var + 1e-5f);
#pragma unroll
    for (int j = 0; j < 3; ++j) {
        const int i = threadIdx.x + j * 256;
        rrow[i] = v[j];
        X[(size_t)t * DMODEL + i] = (v[j] - mu) * rs * w[i] + b[i];
    }
}

// ---------------------------------------------------------------------------
// Causal depthwise conv (DC=4) over xz[:, :DINNER] + bias + SiLU -> xc
// ---------------------------------------------------------------------------
__global__ void conv_silu_kernel(const float* __restrict__ xz,
                                 const float* __restrict__ cw,
                                 const float* __restrict__ cb,
                                 float* __restrict__ xc)
{
    const size_t idx = (size_t)blockIdx.x * blockDim.x + threadIdx.x;
    if (idx >= (size_t)SEQLEN * DINNER) return;
    const int  e = (int)(idx % DINNER);
    const long t = (long)(idx / DINNER);
    float acc = cb[e];
#pragma unroll
    for (int k = 0; k < DCONV; ++k) {
        const long tt = t + k - (DCONV - 1);
        if (tt >= 0)
            acc += cw[e * DCONV + k] * xz[(size_t)tt * (2 * DINNER) + e];
    }
    xc[idx] = siluf(acc);
}

// ---------------------------------------------------------------------------
// Selective scan: one thread per channel d; 16-state recurrence in registers.
// ---------------------------------------------------------------------------
__global__ void scan_kernel(const float* __restrict__ dt,
                            const float* __restrict__ xdbl,
                            const float* __restrict__ xc,
                            const float* __restrict__ A_log,
                            float* __restrict__ ys)
{
    const int d = blockIdx.x * blockDim.x + threadIdx.x;
    if (d >= DINNER) return;
    float A[DSTATE], h[DSTATE];
#pragma unroll
    for (int s = 0; s < DSTATE; ++s) {
        A[s] = -__expf(A_log[(size_t)d * DSTATE + s]);
        h[s] = 0.0f;
    }
    for (int t = 0; t < SEQLEN; ++t) {
        const float dtv = dt[(size_t)t * DINNER + d];
        const float xv  = xc[(size_t)t * DINNER + d];
        const float dx  = dtv * xv;
        const float* xr = xdbl + (size_t)t * XPROJ_N;
        float y = 0.0f;
#pragma unroll
        for (int s = 0; s < DSTATE; ++s) {
            const float dA = __expf(dtv * A[s]);
            h[s] = h[s] * dA + dx * xr[DTRANK + s];
            y += h[s] * xr[DTRANK + DSTATE + s];
        }
        ys[(size_t)t * DINNER + d] = y;
    }
}

// ---------------------------------------------------------------------------
// Gating: ys = (ys + Dp*xc) * silu(z),  z = xz[:, DINNER:]
// ---------------------------------------------------------------------------
__global__ void gated_kernel(float* __restrict__ ys,
                             const float* __restrict__ xc,
                             const float* __restrict__ xz,
                             const float* __restrict__ Dp)
{
    const size_t idx = (size_t)blockIdx.x * blockDim.x + threadIdx.x;
    if (idx >= (size_t)SEQLEN * DINNER) return;
    const int    d = (int)(idx % DINNER);
    const size_t t = idx / DINNER;
    const float  y = ys[idx] + Dp[d] * xc[idx];
    const float  z = xz[t * (2 * DINNER) + DINNER + d];
    ys[idx] = y * siluf(z);
}

// ---------------------------------------------------------------------------
// Launcher
// ---------------------------------------------------------------------------
extern "C" void kernel_launch(void* const* d_in, const int* in_sizes, int n_in,
                              void* d_out, int out_size, void* d_ws, size_t ws_size,
                              hipStream_t stream)
{
    (void)in_sizes; (void)n_in; (void)out_size; (void)ws_size;

    const int*   tokens    = (const int*)  d_in[0];
    const float* embed     = (const float*)d_in[1];
    const float* in_proj_w = (const float*)d_in[2];
    const float* conv_w    = (const float*)d_in[3];
    const float* conv_b    = (const float*)d_in[4];
    const float* x_proj_w  = (const float*)d_in[5];
    const float* dt_proj_w = (const float*)d_in[6];
    const float* dt_proj_b = (const float*)d_in[7];
    const float* A_log     = (const float*)d_in[8];
    const float* Dp        = (const float*)d_in[9];
    const float* out_proj_w= (const float*)d_in[10];
    const float* norm_w    = (const float*)d_in[11];
    const float* norm_b    = (const float*)d_in[12];
    const float* norm_f_w  = (const float*)d_in[13];
    const float* norm_f_b  = (const float*)d_in[14];
    const float* head_w    = (const float*)d_in[15];
    const float* head_b    = (const float*)d_in[16];
    float* logits = (float*)d_out;

    // Workspace layout (floats)
    const size_t LD  = (size_t)SEQLEN * DMODEL;
    const size_t LDI = (size_t)SEQLEN * DINNER;
    float* H  = (float*)d_ws;        // hidden
    float* R  = H  + LD;             // residual
    float* X  = R  + LD;             // post-LN activations
    float* XZ = X  + LD;             // in_proj output [L, 2*DI]
    float* XC = XZ + 2 * LDI;        // conv+silu output [L, DI]
    float* XD = XC + LDI;            // x_proj output [L, 80]
    float* DT = XD + (size_t)SEQLEN * XPROJ_N;  // softplus(dt) [L, DI]
    float* YS = DT + LDI;            // scan output / gated y [L, DI]

    const dim3 blk(256);
    const int ew_blocks = (int)(((size_t)SEQLEN * DINNER + 255) / 256);
    const int mgrid = (SEQLEN + 127) / 128;
    auto ngrid = [](int n) { return (n + 127) / 128; };

    // Embedding
    embed_kernel<<<SEQLEN, blk, 0, stream>>>(tokens, embed, H);

    for (int l = 0; l < NLAYER; ++l) {
        // residual add + pre-norm
        add_ln_kernel<<<SEQLEN, blk, 0, stream>>>(
            H, R, norm_w + (size_t)l * DMODEL, norm_b + (size_t)l * DMODEL,
            X, (l == 0) ? 1 : 0);

        // in_proj: [L,768] x [3072,768]^T -> XZ[L,3072]
        gemm_wmma_kernel<0><<<dim3(ngrid(2 * DINNER), mgrid), blk, 0, stream>>>(
            X, DMODEL,
            in_proj_w + (size_t)l * 2 * DINNER * DMODEL, DMODEL,
            nullptr, XZ, 2 * DINNER, SEQLEN, 2 * DINNER, DMODEL);

        // conv + SiLU -> XC[L,1536]
        conv_silu_kernel<<<ew_blocks, blk, 0, stream>>>(
            XZ, conv_w + (size_t)l * DINNER * DCONV,
            conv_b + (size_t)l * DINNER, XC);

        // x_proj: [L,1536] x [80,1536]^T -> XD[L,80]
        gemm_wmma_kernel<0><<<dim3(ngrid(XPROJ_N), mgrid), blk, 0, stream>>>(
            XC, DINNER,
            x_proj_w + (size_t)l * XPROJ_N * DINNER, DINNER,
            nullptr, XD, XPROJ_N, SEQLEN, XPROJ_N, DINNER);

        // dt_proj (+bias, softplus): [L,48] x [1536,48]^T -> DT[L,1536]
        gemm_wmma_kernel<2><<<dim3(ngrid(DINNER), mgrid), blk, 0, stream>>>(
            XD, XPROJ_N,
            dt_proj_w + (size_t)l * DINNER * DTRANK, DTRANK,
            dt_proj_b + (size_t)l * DINNER, DT, DINNER,
            SEQLEN, DINNER, DTRANK);

        // selective scan -> YS[L,1536]
        scan_kernel<<<(DINNER + 255) / 256, blk, 0, stream>>>(
            DT, XD, XC, A_log + (size_t)l * DINNER * DSTATE, YS);

        // gating (in-place on YS)
        gated_kernel<<<ew_blocks, blk, 0, stream>>>(
            YS, XC, XZ, Dp + (size_t)l * DINNER);

        // out_proj: [L,1536] x [768,1536]^T -> H[L,768]
        gemm_wmma_kernel<0><<<dim3(ngrid(DMODEL), mgrid), blk, 0, stream>>>(
            YS, DINNER,
            out_proj_w + (size_t)l * DMODEL * DINNER, DINNER,
            nullptr, H, DMODEL, SEQLEN, DMODEL, DINNER);
    }

    // final residual + LN
    add_ln_kernel<<<SEQLEN, blk, 0, stream>>>(H, R, norm_f_w, norm_f_b, X, 0);

    // head: [L,768] x [32000,768]^T + bias -> logits[L,32000]
    gemm_wmma_kernel<1><<<dim3(ngrid(V_OUT), mgrid), blk, 0, stream>>>(
        X, DMODEL, head_w, DMODEL, head_b, logits, V_OUT,
        SEQLEN, V_OUT, DMODEL);
}
